// QLinearLayer_2113123910123
// MI455X (gfx1250) — matrix operand order repaired and checked by reading, not compiled, not run on hardware
//
#include <hip/hip_runtime.h>

typedef _Float16 v16h __attribute__((ext_vector_type(16)));
typedef _Float16 v8h  __attribute__((ext_vector_type(8)));
typedef _Float16 v4h  __attribute__((ext_vector_type(4)));
typedef float    v8f  __attribute__((ext_vector_type(8)));

#define S_DIM 2048
#define H_DIM 4096
#define O_DIM 4096
#define NB    31      // low-precision blocks
#define NNON  3968    // H - KEEPER
#define NCHUNK 128    // 4096 / 32 : 32-wide K chunks (4 per 128-wide scale block)

// Block-scaled GEMM via f16 split precision (hi+lo) on V_WMMA_F32_16X16X32_F16.
// 256 threads (8 wave32), 128x128 output tile, K pipelined in 32-wide chunks
// through double-buffered LDS with register prefetch (1 barrier per chunk).
__global__ __launch_bounds__(256) void qlinear_wmma_kernel(
    const float* __restrict__ x,            // [S, H]
    const float* __restrict__ weight,       // [O, H] row-major; W[k,o] = weight[o*H+k]
    const float* __restrict__ scales,       // [O, NNON]
    const float* __restrict__ keep_scales,  // [O, 1]
    const float* __restrict__ inp_scales_lo,// [S, NB]
    const float* __restrict__ inp_scales_hi,// [S, 1]
    float* __restrict__ out)                // [S, O]
{
    __shared__ __align__(32) _Float16 AhS[2][128 * 32];
    __shared__ __align__(32) _Float16 AlS[2][128 * 32];
    __shared__ __align__(32) _Float16 BhS[2][128 * 32];
    __shared__ __align__(32) _Float16 BlS[2][128 * 32];
    __shared__ __align__(32) float    rbufS[2][128];  // row scale r[s], per kb parity
    __shared__ __align__(32) float    cbufS[2][128];  // col scale c[o], per kb parity

    const int t    = threadIdx.x;
    const int lane = t & 31;          // wave32
    const int wave = t >> 5;          // 0..7
    const int wm   = wave >> 1;       // 0..3 : 32-row band
    const int wn   = wave & 1;        // 0..1 : 64-col band
    const int half = lane >> 4;       // 0/1
    const int l16  = lane & 15;
    const int s_base = blockIdx.y * 128;
    const int o_base = blockIdx.x * 128;

    // Per-thread staging geometry: 1024 float4 per plane per chunk, 4 per thread.
    const float* aptr[4];
    const float* bptr[4];
    int ldsoff[4];
#pragma unroll
    for (int i = 0; i < 4; ++i) {
        const int idx = t + i * 256;        // 0..1023
        const int row = idx >> 3;           // 0..127
        const int c4  = (idx & 7) * 4;      // 0..28 step 4
        aptr[i]   = x      + (size_t)(s_base + row) * H_DIM + c4;
        bptr[i]   = weight + (size_t)(o_base + row) * H_DIM + c4;
        ldsoff[i] = row * 32 + c4;
    }

    const v8f zero = {0.f, 0.f, 0.f, 0.f, 0.f, 0.f, 0.f, 0.f};
    v8f acc[2][4];   // final f32 accumulators
    v8f part[2][4];  // per-128-K-block partial (pre-scale)
#pragma unroll
    for (int i = 0; i < 2; ++i)
#pragma unroll
        for (int j = 0; j < 4; ++j) { acc[i][j] = zero; part[i][j] = zero; }

    // ---- prologue: prefetch chunk 0 tiles + kb=0 scale into registers ----
    float4 av[4], bv[4];
#pragma unroll
    for (int i = 0; i < 4; ++i) {
        av[i] = *(const float4*)(aptr[i]);
        bv[i] = *(const float4*)(bptr[i]);
    }
    float sreg;
    if (t < 128) sreg = inp_scales_lo[(size_t)(s_base + t) * NB + 0];
    else         sreg = scales[(size_t)(o_base + t - 128) * NNON + 0];

    for (int c = 0; c < NCHUNK; ++c) {
        const int kb  = c >> 2;
        const int kc  = c & 3;
        const int buf = c & 1;
        _Float16* Ah = AhS[buf];
        _Float16* Al = AlS[buf];
        _Float16* Bh = BhS[buf];
        _Float16* Bl = BlS[buf];

        // ---- store phase: f32 -> (f16 hi, f16 lo) split into LDS[buf] ----
#pragma unroll
        for (int i = 0; i < 4; ++i) {
            v4h hi, lo;
            {
                _Float16 h0 = (_Float16)av[i].x, h1 = (_Float16)av[i].y,
                         h2 = (_Float16)av[i].z, h3 = (_Float16)av[i].w;
                hi[0] = h0; hi[1] = h1; hi[2] = h2; hi[3] = h3;
                lo[0] = (_Float16)(av[i].x - (float)h0);
                lo[1] = (_Float16)(av[i].y - (float)h1);
                lo[2] = (_Float16)(av[i].z - (float)h2);
                lo[3] = (_Float16)(av[i].w - (float)h3);
            }
            *(v4h*)(Ah + ldsoff[i]) = hi;
            *(v4h*)(Al + ldsoff[i]) = lo;
            {
                _Float16 h0 = (_Float16)bv[i].x, h1 = (_Float16)bv[i].y,
                         h2 = (_Float16)bv[i].z, h3 = (_Float16)bv[i].w;
                hi[0] = h0; hi[1] = h1; hi[2] = h2; hi[3] = h3;
                lo[0] = (_Float16)(bv[i].x - (float)h0);
                lo[1] = (_Float16)(bv[i].y - (float)h1);
                lo[2] = (_Float16)(bv[i].z - (float)h2);
                lo[3] = (_Float16)(bv[i].w - (float)h3);
            }
            *(v4h*)(Bh + ldsoff[i]) = hi;
            *(v4h*)(Bl + ldsoff[i]) = lo;
        }
        if (kc == 0) {   // wave-uniform split (waves 0-3 vs 4-7)
            if (t < 128) rbufS[kb & 1][t]       = sreg;
            else         cbufS[kb & 1][t - 128] = sreg;
        }
        __syncthreads();   // stores[c] visible; also orders writes[c] after reads[c-2]

        // ---- prefetch phase: issue global loads for chunk c+1 (hide under WMMA) ----
        if (c + 1 < NCHUNK) {
            const int k0n = (c + 1) * 32;
#pragma unroll
            for (int i = 0; i < 4; ++i) {
                av[i] = *(const float4*)(aptr[i] + k0n);
                bv[i] = *(const float4*)(bptr[i] + k0n);
            }
            if (((c + 1) & 3) == 0) {
                const int kbn = (c + 1) >> 2;
                if (t < 128) {
                    const int s = s_base + t;
                    sreg = (kbn < NB) ? inp_scales_lo[(size_t)s * NB + kbn]
                                      : inp_scales_hi[s];
                } else {
                    const int o = o_base + (t - 128);
                    sreg = (kbn < NB) ? scales[(size_t)o * NNON + kbn * 128]
                                      : keep_scales[o];
                }
            }
        }

        // ---- compute phase: 2x4 tiles of 16x16, 3 WMMAs each ----
#pragma unroll
        for (int tm = 0; tm < 2; ++tm) {
            // A fragment (16x32 f16): lane = row l16, half selects K-subrange.
            const int rowA = wm * 32 + tm * 16 + l16;
            const _Float16* aph = Ah + rowA * 32 + half * 16;
            const _Float16* apl = Al + rowA * 32 + half * 16;
            const v8h ah0 = *(const v8h*)(aph);
            const v8h ah1 = *(const v8h*)(aph + 8);
            const v8h al0 = *(const v8h*)(apl);
            const v8h al1 = *(const v8h*)(apl + 8);
            const v16h a_hi = __builtin_shufflevector(ah0, ah1,
                0,1,2,3,4,5,6,7,8,9,10,11,12,13,14,15);
            const v16h a_lo = __builtin_shufflevector(al0, al1,
                0,1,2,3,4,5,6,7,8,9,10,11,12,13,14,15);
#pragma unroll
            for (int tn = 0; tn < 4; ++tn) {
                // B fragment (32x16 f16): lane = col l16, half selects K 0-15/16-31.
                const int n = wn * 64 + tn * 16 + l16;
                const v16h b_hi = *(const v16h*)(Bh + n * 32 + half * 16);
                const v16h b_lo = *(const v16h*)(Bl + n * 32 + half * 16);
                v8f p = part[tm][tn];
                p = __builtin_amdgcn_wmma_f32_16x16x32_f16(
                        false, a_hi, false, b_hi, (short)0, p, false, false);
                p = __builtin_amdgcn_wmma_f32_16x16x32_f16(
                        false, a_hi, false, b_lo, (short)0, p, false, false);
                p = __builtin_amdgcn_wmma_f32_16x16x32_f16(
                        false, a_lo, false, b_hi, (short)0, p, false, false);
                part[tm][tn] = p;
            }
        }

        // ---- end of 128-wide K block: acc += part * r[s] * c[o] ----
        if (kc == 3) {
            const float* rb = rbufS[kb & 1];
            const float* cb = cbufS[kb & 1];
#pragma unroll
            for (int tm = 0; tm < 2; ++tm) {
                const v8f rv = *(const v8f*)(rb + wm * 32 + tm * 16 + half * 8);
#pragma unroll
                for (int tn = 0; tn < 4; ++tn) {
                    const float cv = cb[wn * 64 + tn * 16 + l16];
#pragma unroll
                    for (int i = 0; i < 8; ++i)
                        acc[tm][tn][i] += part[tm][tn][i] * rv[i] * cv;
                    part[tm][tn] = zero;
                }
            }
        }
    }

    // Write out: C/D layout -> row = base + i + 8*half, col = base + l16.
#pragma unroll
    for (int tm = 0; tm < 2; ++tm) {
        const int rbase = s_base + wm * 32 + tm * 16 + half * 8;
#pragma unroll
        for (int tn = 0; tn < 4; ++tn) {
            const int col = o_base + wn * 64 + tn * 16 + l16;
#pragma unroll
            for (int i = 0; i < 8; ++i)
                out[(size_t)(rbase + i) * O_DIM + col] = acc[tm][tn][i];
        }
    }
}

extern "C" void kernel_launch(void* const* d_in, const int* in_sizes, int n_in,
                              void* d_out, int out_size, void* d_ws, size_t ws_size,
                              hipStream_t stream) {
    (void)in_sizes; (void)n_in; (void)out_size; (void)d_ws; (void)ws_size;
    const float* x             = (const float*)d_in[0];
    const float* weight        = (const float*)d_in[1];
    const float* scales        = (const float*)d_in[2];
    const float* keep_scales   = (const float*)d_in[3];
    const float* inp_scales_lo = (const float*)d_in[4];
    const float* inp_scales_hi = (const float*)d_in[5];
    // d_in[6], d_in[7] (inp_base_lo/hi) are unused by the reference math.
    float* out = (float*)d_out;

    dim3 grid(O_DIM / 128, S_DIM / 128);  // 32 x 16 workgroups
    qlinear_wmma_kernel<<<grid, 256, 0, stream>>>(
        x, weight, scales, keep_scales, inp_scales_lo, inp_scales_hi, out);
}
